// MultiHeadAttention_48172353192050
// MI455X (gfx1250) — matrix operand order repaired and checked by reading, not compiled, not run on hardware
//
#include <hip/hip_runtime.h>
#include <hip/hip_bf16.h>
#include <stdint.h>
#include <stddef.h>

// Problem dims (fixed by the reference)
#define SEQ   4096
#define DIN   1024
#define DOUTD 1024
#define NH    16
#define HDIM  64

typedef __bf16 bf16_t;
typedef __attribute__((ext_vector_type(16))) __bf16 v16bf;
typedef __attribute__((ext_vector_type(8)))  float  v8f;
typedef __attribute__((ext_vector_type(4)))  uint32_t u32x4;
typedef __attribute__((ext_vector_type(8)))  uint32_t u32x8;

// ---------------------------------------------------------------------------
// WMMA helpers (CDNA5 layouts, wave32)
// ---------------------------------------------------------------------------
__device__ __forceinline__ v8f wmma_bf16(v16bf a, v16bf b, v8f c) {
  // D = A(16x32 bf16) * B(32x16 bf16) + C(16x16 f32)
  return __builtin_amdgcn_wmma_f32_16x16x32_bf16(
      /*neg_a=*/false, a, /*neg_b=*/false, b,
      /*c_mod=*/(short)0, c, /*reuse_a=*/false, /*reuse_b=*/false);
}

// A-matrix (16x32, bf16) load from row-major [M][>=32] with leading dim ld.
// ISA 7.12.2: lane L holds row m=L%16; VGPR v holds K=(v/4)*16+(L/16)*8+(v%4)*2.
__device__ __forceinline__ v16bf load_A_rm(const bf16_t* p, int ld) {
  const int lane = threadIdx.x & 31;
  const int m    = lane & 15;
  const int half = lane >> 4;
  v16bf a;
#pragma unroll
  for (int v = 0; v < 8; ++v) {
    const int kb = ((v >> 2) << 4) + (half << 3) + ((v & 3) << 1);
    a[2 * v]     = p[m * ld + kb];
    a[2 * v + 1] = p[m * ld + kb + 1];
  }
  return a;
}

// B-matrix (32x16, bf16): element B[k][n] = p[n*ld + k] (p = row-major [N][K]
// tile). Lane L holds column n=L%16; VGPR v holds
// K=(v/4)*16+(L/16)*8+(v%4)*2 (same order as load_A_rm).
__device__ __forceinline__ v16bf load_B_nk(const bf16_t* p, int ld) {
  const int lane = threadIdx.x & 31;
  const int n    = lane & 15;
  const int half = lane >> 4;
  v16bf b;
#pragma unroll
  for (int v = 0; v < 8; ++v) {
    const int k = ((v >> 2) << 4) + (half << 3) + ((v & 3) << 1);
    b[2 * v]     = p[n * ld + k];
    b[2 * v + 1] = p[n * ld + k + 1];
  }
  return b;
}

// ---------------------------------------------------------------------------
// Tensor Data Mover: stage a 2D bf16 tile (rows x cols, row stride in elems)
// from global memory into LDS. D# per CDNA5 ISA.
// Wave-uniform operands; EXEC is ignored by TDM. Tracked by TENSORcnt.
// ---------------------------------------------------------------------------
__device__ __forceinline__ void tdm_load_2d_bf16(const bf16_t* gsrc,
                                                 const bf16_t* lds_dst,
                                                 uint32_t cols, uint32_t rows,
                                                 uint32_t stride) {
  const uint64_t ga = (uint64_t)(uintptr_t)gsrc;
  const uint32_t la = (uint32_t)(uintptr_t)lds_dst;  // addr[31:0] -> LDS offset
  u32x4 g0;
  g0[0] = 1u;                                   // count=1 (valid user D#)
  g0[1] = la;                                   // lds_addr [63:32]
  g0[2] = (uint32_t)ga;                         // global_addr[31:0]
  g0[3] = (uint32_t)((ga >> 32) & 0x01FFFFFFu)  // global_addr[56:32]
          | (2u << 30);                         // type=2 ("image") [127:126]
  u32x8 g1;
  g1[0] = 1u << 16;                             // data_size=1 (2B); wg_mask=0
  g1[1] = (cols & 0xFFFFu) << 16;               // tensor_dim0[15:0] @ [63:48]
  g1[2] = ((cols >> 16) & 0xFFFFu)              // tensor_dim0[31:16]
          | ((rows & 0xFFFFu) << 16);           // tensor_dim1[15:0]
  g1[3] = ((rows >> 16) & 0xFFFFu)              // tensor_dim1[31:16]
          | ((cols & 0xFFFFu) << 16);           // tile_dim0
  g1[4] = rows & 0xFFFFu;                       // tile_dim1 (tile_dim2 = 0)
  g1[5] = stride;                               // tensor_dim0_stride[31:0]
  g1[6] = 0u;                                   // stride[47:32]=0, d1stride lo
  g1[7] = 0u;                                   // d1stride hi (unused, 2D)
  asm volatile("tensor_load_to_lds %0, %1" :: "s"(g0), "s"(g1) : "memory");
}

// ---------------------------------------------------------------------------
// Kernel 0a: f32 -> bf16 cast (x)
// ---------------------------------------------------------------------------
__global__ void mha_cast_x(const float* __restrict__ x, bf16_t* __restrict__ xb,
                           int n) {
  int i = blockIdx.x * blockDim.x + threadIdx.x;
  if (i < n) xb[i] = (bf16_t)x[i];
}

// Kernel 0b: f32 [K][N] -> bf16 transposed [N][K]  (K = N = 1024)
__global__ void mha_cast_wt(const float* __restrict__ W,
                            bf16_t* __restrict__ Wt) {
  int i = blockIdx.x * blockDim.x + threadIdx.x;  // over 1024*1024
  int nidx = i >> 10;
  int k    = i & 1023;
  Wt[i] = (bf16_t)W[k * 1024 + nidx];  // Wt[n][k] = W[k][n]
}

// ---------------------------------------------------------------------------
// Kernel 1: projection GEMM  C(4096x1024) = xb @ W.  One wave = 32x64 tile
// (8 WMMA per K-step against 12 b128 loads). Output head-major bf16:
//   modeT=0 -> [H][S][HD] (Q, K);  modeT=1 -> [H][HD][S] (V, pre-transposed)
// ---------------------------------------------------------------------------
__global__ void mha_qkv_gemm(const bf16_t* __restrict__ xb,
                             const bf16_t* __restrict__ Wt,  // [DOUT][DIN]
                             bf16_t* __restrict__ dst, int modeT) {
  const int row0 = blockIdx.x * 32;
  const int col0 = blockIdx.y * 64;
  v8f acc[2][4] = {};
#pragma unroll 2
  for (int k0 = 0; k0 < DIN; k0 += 32) {
    __builtin_prefetch(xb + (row0) * DIN + k0 + 128, 0, 0);
    __builtin_prefetch(Wt + (col0) * DIN + k0 + 128, 0, 0);
    v16bf A[2], B[4];
    A[0] = load_A_rm(xb + (size_t)(row0 + 0)  * DIN + k0, DIN);
    A[1] = load_A_rm(xb + (size_t)(row0 + 16) * DIN + k0, DIN);
#pragma unroll
    for (int j = 0; j < 4; ++j)
      B[j] = load_B_nk(Wt + (size_t)(col0 + 16 * j) * DIN + k0, DIN);
#pragma unroll
    for (int i = 0; i < 2; ++i)
#pragma unroll
      for (int j = 0; j < 4; ++j)
        acc[i][j] = wmma_bf16(A[i], B[j], acc[i][j]);
  }
  const int lane = threadIdx.x & 31;
  const int n    = lane & 15;
  const int half = lane >> 4;
#pragma unroll
  for (int i = 0; i < 2; ++i)
#pragma unroll
    for (int j = 0; j < 4; ++j)
#pragma unroll
      for (int r = 0; r < 8; ++r) {
        const int m = row0 + i * 16 + half * 8 + r;
        const int c = col0 + j * 16 + n;
        const bf16_t bv = (bf16_t)(acc[i][j][r]);
        const int h = c >> 6, d = c & 63;
        if (modeT)
          dst[(size_t)(h * HDIM + d) * SEQ + m] = bv;   // [H][HD][S]
        else
          dst[((size_t)h * SEQ + m) * HDIM + d] = bv;   // [H][S][HD]
      }
}

// ---------------------------------------------------------------------------
// Kernel 2: causal flash attention. One wave per (head, 16-query tile).
// K/V 32-key blocks are staged into LDS by the Tensor Data Mover, double
// buffered so the TDM for block kb+1 overlaps WMMA on block kb. Scores via
// WMMA bf16, online softmax across the 16-lane row groups, P converted
// C-layout -> A-layout through LDS, PV via WMMA.
// ---------------------------------------------------------------------------
__global__ void mha_flash_attn(const bf16_t* __restrict__ Qh,
                               const bf16_t* __restrict__ Kh,
                               const bf16_t* __restrict__ Vt,
                               bf16_t* __restrict__ ctx) {
  __shared__ bf16_t kbuf[2][32 * HDIM];   // K block: 32 keys x 64 dims
  __shared__ bf16_t vbuf[2][HDIM * 32];   // V block: 64 dims x 32 keys
  __shared__ bf16_t plds[16 * 32];        // P tile for C->A relayout

  const int qt = blockIdx.x;   // 0..SEQ/16-1
  const int h  = blockIdx.y;   // 0..NH-1
  const int q0 = qt * 16;
  const int lane = threadIdx.x & 31;
  const int n    = lane & 15;
  const int half = lane >> 4;

  const bf16_t* Qb = Qh + (size_t)h * SEQ * HDIM;
  const bf16_t* Kb = Kh + (size_t)h * SEQ * HDIM;
  const bf16_t* Vb = Vt + (size_t)h * HDIM * SEQ;

  // Query tile, split along K: qa0 = d 0..31, qa1 = d 32..63
  const v16bf qa0 = load_A_rm(Qb + (size_t)q0 * HDIM, HDIM);
  const v16bf qa1 = load_A_rm(Qb + (size_t)q0 * HDIM + 32, HDIM);

  v8f o0 = {}, o1 = {}, o2 = {}, o3 = {};
  float mrow[8], lrow[8];
#pragma unroll
  for (int r = 0; r < 8; ++r) { mrow[r] = -1e30f; lrow[r] = 0.0f; }

  const float scale = 0.125f;             // 1/sqrt(64)
  const int nblk = (q0 + 16 + 31) / 32;   // causal: key blocks of 32

  // Prologue: TDM-stage block 0 (K then V) into buffer 0.
  tdm_load_2d_bf16(Kb, kbuf[0], /*cols=*/HDIM, /*rows=*/32, /*stride=*/HDIM);
  tdm_load_2d_bf16(Vb, vbuf[0], /*cols=*/32, /*rows=*/HDIM, /*stride=*/SEQ);

  for (int kb = 0; kb < nblk; ++kb) {
    const int buf = kb & 1;
    const int key0 = kb * 32;
    if (kb + 1 < nblk) {
      // WAR guard: in-flight ds reads of the buffer we are about to refill
      asm volatile("s_wait_dscnt 0x0" ::: "memory");
      tdm_load_2d_bf16(Kb + (size_t)(key0 + 32) * HDIM, kbuf[buf ^ 1],
                       HDIM, 32, HDIM);
      tdm_load_2d_bf16(Vb + (size_t)(key0 + 32), vbuf[buf ^ 1],
                       32, HDIM, SEQ);
      // 2 outstanding TDMs allowed (the prefetch pair); current pair done.
      asm volatile("s_wait_tensorcnt 0x2" ::: "memory");
    } else {
      asm volatile("s_wait_tensorcnt 0x0" ::: "memory");
    }

    const bf16_t* kblk = kbuf[buf];
    const bf16_t* vblk = vbuf[buf];

    // ---- scores: S(16x32) = Q(16x64) @ K^T (from LDS) ----
    v8f c0 = {}, c1 = {};
    c0 = wmma_bf16(qa0, load_B_nk(kblk, HDIM), c0);
    c0 = wmma_bf16(qa1, load_B_nk(kblk + 32, HDIM), c0);
    c1 = wmma_bf16(qa0, load_B_nk(kblk + 16 * HDIM, HDIM), c1);
    c1 = wmma_bf16(qa1, load_B_nk(kblk + 16 * HDIM + 32, HDIM), c1);

    // ---- mask + scale + online softmax ----
#pragma unroll
    for (int r = 0; r < 8; ++r) {
      const int q = q0 + half * 8 + r;
      float s0 = (key0 + n      <= q) ? c0[r] * scale : -1e30f;
      float s1 = (key0 + 16 + n <= q) ? c1[r] * scale : -1e30f;
      float bm = fmaxf(s0, s1);
#pragma unroll
      for (int msk = 1; msk < 16; msk <<= 1)
        bm = fmaxf(bm, __shfl_xor(bm, msk, 32));
      const float mnew  = fmaxf(mrow[r], bm);
      const float alpha = __expf(mrow[r] - mnew);
      const float p0 = __expf(s0 - mnew);
      const float p1 = __expf(s1 - mnew);
      float ps = p0 + p1;
#pragma unroll
      for (int msk = 1; msk < 16; msk <<= 1) ps += __shfl_xor(ps, msk, 32);
      lrow[r] = lrow[r] * alpha + ps;
      mrow[r] = mnew;
      o0[r] *= alpha; o1[r] *= alpha; o2[r] *= alpha; o3[r] *= alpha;
      const int m = half * 8 + r;
      plds[m * 32 + n]      = (bf16_t)p0;   // P row-major 16x32 in LDS
      plds[m * 32 + 16 + n] = (bf16_t)p1;
    }
    // cross-lane LDS RAW within the wave
    asm volatile("s_wait_dscnt 0x0" ::: "memory");

    const v16bf pa = load_A_rm(plds, 32);   // P as A-matrix (16x32 bf16)

    // ---- O += P @ V  (V block in LDS as [HD][32]) ----
    o0 = wmma_bf16(pa, load_B_nk(vblk + 0  * 32, 32), o0);
    o1 = wmma_bf16(pa, load_B_nk(vblk + 16 * 32, 32), o1);
    o2 = wmma_bf16(pa, load_B_nk(vblk + 32 * 32, 32), o2);
    o3 = wmma_bf16(pa, load_B_nk(vblk + 48 * 32, 32), o3);
  }

  // ---- finalize: divide by row sums, scatter into ctx [S][DOUT] bf16 ----
#pragma unroll
  for (int r = 0; r < 8; ++r) {
    const int m = q0 + half * 8 + r;
    const float inv = 1.0f / lrow[r];
    bf16_t* row = ctx + (size_t)m * DOUTD + h * HDIM;
    row[0  + n] = (bf16_t)(o0[r] * inv);
    row[16 + n] = (bf16_t)(o1[r] * inv);
    row[32 + n] = (bf16_t)(o2[r] * inv);
    row[48 + n] = (bf16_t)(o3[r] * inv);
  }
}

// ---------------------------------------------------------------------------
// Kernel 3: output projection  out = ctx @ Wo + bo  (f32 out), 32x64/wave
// ---------------------------------------------------------------------------
__global__ void mha_out_gemm(const bf16_t* __restrict__ ctx,
                             const bf16_t* __restrict__ Wot,  // [DOUT][DOUT]
                             const float* __restrict__ bo,
                             float* __restrict__ out) {
  const int row0 = blockIdx.x * 32;
  const int col0 = blockIdx.y * 64;
  v8f acc[2][4] = {};
#pragma unroll 2
  for (int k0 = 0; k0 < DOUTD; k0 += 32) {
    __builtin_prefetch(ctx + (row0) * DOUTD + k0 + 128, 0, 0);
    __builtin_prefetch(Wot + (col0) * DOUTD + k0 + 128, 0, 0);
    v16bf A[2], B[4];
    A[0] = load_A_rm(ctx + (size_t)(row0 + 0)  * DOUTD + k0, DOUTD);
    A[1] = load_A_rm(ctx + (size_t)(row0 + 16) * DOUTD + k0, DOUTD);
#pragma unroll
    for (int j = 0; j < 4; ++j)
      B[j] = load_B_nk(Wot + (size_t)(col0 + 16 * j) * DOUTD + k0, DOUTD);
#pragma unroll
    for (int i = 0; i < 2; ++i)
#pragma unroll
      for (int j = 0; j < 4; ++j)
        acc[i][j] = wmma_bf16(A[i], B[j], acc[i][j]);
  }
  const int lane = threadIdx.x & 31;
  const int n    = lane & 15;
  const int half = lane >> 4;
#pragma unroll
  for (int i = 0; i < 2; ++i)
#pragma unroll
    for (int j = 0; j < 4; ++j)
#pragma unroll
      for (int r = 0; r < 8; ++r) {
        const int m = row0 + i * 16 + half * 8 + r;
        const int c = col0 + j * 16 + n;
        out[(size_t)m * DOUTD + c] = acc[i][j][r] + bo[c];
      }
}

// ---------------------------------------------------------------------------
// Host-side launcher
// ---------------------------------------------------------------------------
extern "C" void kernel_launch(void* const* d_in, const int* in_sizes, int n_in,
                              void* d_out, int out_size, void* d_ws,
                              size_t ws_size, hipStream_t stream) {
  const float* x  = (const float*)d_in[0];
  const float* Wq = (const float*)d_in[1];
  const float* Wk = (const float*)d_in[2];
  const float* Wv = (const float*)d_in[3];
  const float* Wo = (const float*)d_in[4];
  const float* bo = (const float*)d_in[5];
  float* out = (float*)d_out;

  // Workspace layout (48 MiB total)
  char* ws = (char*)d_ws;
  const size_t MB = 1024ull * 1024ull;
  bf16_t* xb  = (bf16_t*)(ws + 0 * MB);   // [S][DIN]        8 MiB
  bf16_t* Wqt = (bf16_t*)(ws + 8 * MB);   // [DOUT][DIN]     2 MiB
  bf16_t* Wkt = (bf16_t*)(ws + 10 * MB);
  bf16_t* Wvt = (bf16_t*)(ws + 12 * MB);
  bf16_t* Wot = (bf16_t*)(ws + 14 * MB);
  bf16_t* Qh  = (bf16_t*)(ws + 16 * MB);  // [H][S][HD]      8 MiB
  bf16_t* Kh  = (bf16_t*)(ws + 24 * MB);  // [H][S][HD]      8 MiB
  bf16_t* Vt  = (bf16_t*)(ws + 32 * MB);  // [H][HD][S]      8 MiB
  bf16_t* ctx = (bf16_t*)(ws + 40 * MB);  // [S][DOUT]       8 MiB

  // 0) casts
  mha_cast_x<<<(SEQ * DIN + 255) / 256, 256, 0, stream>>>(x, xb, SEQ * DIN);
  mha_cast_wt<<<(1024 * 1024) / 256, 256, 0, stream>>>(Wq, Wqt);
  mha_cast_wt<<<(1024 * 1024) / 256, 256, 0, stream>>>(Wk, Wkt);
  mha_cast_wt<<<(1024 * 1024) / 256, 256, 0, stream>>>(Wv, Wvt);
  mha_cast_wt<<<(1024 * 1024) / 256, 256, 0, stream>>>(Wo, Wot);

  // 1) QKV projections (one wave = 32x64 tile)
  dim3 g1(SEQ / 32, DOUTD / 64);
  mha_qkv_gemm<<<g1, 32, 0, stream>>>(xb, Wqt, Qh, 0);
  mha_qkv_gemm<<<g1, 32, 0, stream>>>(xb, Wkt, Kh, 0);
  mha_qkv_gemm<<<g1, 32, 0, stream>>>(xb, Wvt, Vt, 1);

  // 2) causal flash attention (one wave per head x 16-query tile, TDM-fed)
  mha_flash_attn<<<dim3(SEQ / 16, NH), 32, 0, stream>>>(Qh, Kh, Vt, ctx);

  // 3) output projection + bias
  mha_out_gemm<<<g1, 32, 0, stream>>>(ctx, Wot, bo, out);
}
